// Graphormer_40492951667288
// MI455X (gfx1250) — matrix operand order repaired and checked by reading, not compiled, not run on hardware
//
#include <hip/hip_runtime.h>
#include <hip/hip_bf16.h>

// ---------------------------------------------------------------------------
// Graphormer forward for MI455X (gfx1250, wave32, WMMA).
// All dense math via v_wmma_f32_16x16x32_bf16 (bf16 A/B, f32 accumulate).
// LDS tiles are stored in fragment-native layout so every WMMA operand is
// fetched with two ds_load_b128 per lane. Attention is flash-style with the
// Graphormer spd/edge biases fused; K/V tiles are staged in LDS and shared
// by 4 query-tile waves per block.
// ---------------------------------------------------------------------------

#define NTOK  2048
#define DMODEL 512
#define HEADS 8
#define HDIM  64
#define FFDIM 2048
#define INDIM 1433
#define OUTC  7

typedef __attribute__((ext_vector_type(16))) __bf16 v16bf;
typedef __attribute__((ext_vector_type(4)))  __bf16 v4bf;
typedef __attribute__((ext_vector_type(8)))  float  v8f;

__device__ __forceinline__ int clampi(int v, int lo, int hi) {
    return v < lo ? lo : (v > hi ? hi : v);
}

// A-fragment K permutation: within a 32-wide K chunk store order
// {0..7,16..23,8..15,24..31} so lane-half h reads 16 contiguous bf16.
__device__ __forceinline__ int apermute(int k) {
    return (((k >> 3) & 1) << 4) + (((k >> 4) & 1) << 3) + (k & 7);
}

// ---------------------------------------------------------------------------
// GEMM: C[M,N] = act( A[M,K] @ B[K,N] + bias + emb1[idx1] + emb2[idx2] + res )
// 64x64 tile / 256-thread block (8 waves; each wave owns 16 rows x 32 cols).
// M, N must be multiples of 64 (true for all call sites); K may be ragged.
// ---------------------------------------------------------------------------
__global__ __launch_bounds__(256)
void gemm_bf16_kernel(const float* __restrict__ A, const float* __restrict__ B,
                      const float* __restrict__ bias, float* __restrict__ C,
                      const float* __restrict__ res,
                      const int* __restrict__ idx1, const float* __restrict__ emb1,
                      const int* __restrict__ idx2, const float* __restrict__ emb2,
                      int M, int N, int K, int act)
{
    __shared__ alignas(16) __bf16 As[64][32];   // [row][k]
    __shared__ alignas(16) __bf16 Bs[64][32];   // [n][k]  (transposed tile)

    const int m0 = blockIdx.y * 64;
    const int n0 = blockIdx.x * 64;
    const int tid  = threadIdx.x;
    const int wave = tid >> 5;
    const int lane = tid & 31;
    const int wr = wave >> 1;      // 0..3 : 16-row strip
    const int wc = wave & 1;       // 0..1 : 32-col strip
    const int col  = lane & 15;
    const int half = lane >> 4;

    const bool kvec = ((K & 3) == 0);
    v8f c0 = {}, c1 = {};

    for (int k0 = 0; k0 < K; k0 += 32) {
        if (k0 + 32 < K) {
            __builtin_prefetch(&A[(size_t)(m0 + (tid >> 2)) * K + k0 + 32 + ((tid & 3) << 3)], 0, 1);
            __builtin_prefetch(&B[(size_t)(k0 + 32 + (tid >> 6)) * N + n0 + (tid & 63)], 0, 1);
        }
        const bool kfull = (k0 + 32 <= K);
        if (kfull && kvec) {
            // vectorized staging: global_load_b128 + packed LDS stores
            #pragma unroll
            for (int t = 0; t < 2; ++t) {
                int g = tid + t * 256;                 // 512 groups of 4
                // A: row r, k-group kg (4 contiguous k)
                int r = g >> 3, kg = (g & 7) << 2;
                const float4 va = *reinterpret_cast<const float4*>(
                    &A[(size_t)(m0 + r) * K + k0 + kg]);
                v4bf wa;
                wa[0] = (__bf16)va.x; wa[1] = (__bf16)va.y;
                wa[2] = (__bf16)va.z; wa[3] = (__bf16)va.w;
                *reinterpret_cast<v4bf*>(&As[r][kg]) = wa;
                // B: k-row rk, n-group ng (4 contiguous n) -> transposed store
                int rk = g >> 4, ng = (g & 15) << 2;
                const float4 vb = *reinterpret_cast<const float4*>(
                    &B[(size_t)(k0 + rk) * N + n0 + ng]);
                Bs[ng + 0][rk] = (__bf16)vb.x;
                Bs[ng + 1][rk] = (__bf16)vb.y;
                Bs[ng + 2][rk] = (__bf16)vb.z;
                Bs[ng + 3][rk] = (__bf16)vb.w;
            }
        } else {
            // scalar staging with K-tail guard (embed GEMM / last chunk)
            #pragma unroll
            for (int e = tid; e < 64 * 32; e += 256) {
                int r = e >> 5, cc = e & 31;
                int gk = k0 + cc;
                float va = (gk < K) ? A[(size_t)(m0 + r) * K + gk] : 0.0f;
                As[r][cc] = (__bf16)va;
            }
            #pragma unroll
            for (int e = tid; e < 32 * 64; e += 256) {
                int rk = e >> 6, nn = e & 63;
                int gk = k0 + rk;
                float vb = (gk < K) ? B[(size_t)gk * N + n0 + nn] : 0.0f;
                Bs[nn][rk] = (__bf16)vb;
            }
        }
        __syncthreads();

        // fragment loads: two ds_load_b128 each
        const v16bf a  = reinterpret_cast<const v16bf*>(&As[wr * 16 + col][0])[half];
        const v16bf b0 = reinterpret_cast<const v16bf*>(&Bs[wc * 32 + col][0])[half];
        const v16bf b1 = reinterpret_cast<const v16bf*>(&Bs[wc * 32 + 16 + col][0])[half];

        c0 = __builtin_amdgcn_wmma_f32_16x16x32_bf16(false, a, false, b0, (short)0, c0, false, false);
        c1 = __builtin_amdgcn_wmma_f32_16x16x32_bf16(false, a, false, b1, (short)0, c1, false, false);
        __syncthreads();
    }

    // epilogue: C layout -> element r is row (half*8 + r), col = lane&15
    const int rb = half * 8;
    #pragma unroll
    for (int r = 0; r < 8; ++r) {
        int m = m0 + wr * 16 + rb + r;
        #pragma unroll
        for (int f = 0; f < 2; ++f) {
            int n = n0 + wc * 32 + f * 16 + col;
            float vv = (f == 0) ? c0[r] : c1[r];
            if (bias) vv += bias[n];
            if (emb1) { int i1 = clampi(idx1[m], 0, 50); vv += emb1[(size_t)i1 * N + n]; }
            if (emb2) { int i2 = clampi(idx2[m], 0, 50); vv += emb2[(size_t)i2 * N + n]; }
            if (res)  vv += res[(size_t)m * N + n];
            if (act == 1) vv = 0.5f * vv * (1.0f + erff(vv * 0.70710678118654752f));
            C[(size_t)m * N + n] = vv;
        }
    }
}

// ---------------------------------------------------------------------------
// xc = x + cinE[cin[row]] + coutE[cout[row]]  (float4 per thread)
// ---------------------------------------------------------------------------
__global__ __launch_bounds__(256)
void add_cent_kernel(const float* __restrict__ x,
                     const int* __restrict__ cin, const int* __restrict__ cout,
                     const float* __restrict__ cinE, const float* __restrict__ coutE,
                     float* __restrict__ out)
{
    int idx = (blockIdx.x * 256 + threadIdx.x) * 4;    // NTOK*DMODEL total
    int m = idx >> 9, n = idx & 511;
    int a = clampi(cin[m], 0, 50);
    int c = clampi(cout[m], 0, 50);
    float4 xv = *reinterpret_cast<const float4*>(&x[idx]);
    float4 av = *reinterpret_cast<const float4*>(&cinE[a * DMODEL + n]);
    float4 cv = *reinterpret_cast<const float4*>(&coutE[c * DMODEL + n]);
    float4 o;
    o.x = xv.x + av.x + cv.x; o.y = xv.y + av.y + cv.y;
    o.z = xv.z + av.z + cv.z; o.w = xv.w + av.w + cv.w;
    *reinterpret_cast<float4*>(&out[idx]) = o;
}

// ---------------------------------------------------------------------------
// Flash attention with Graphormer biases.
// Block = 128 threads (4 waves) = 64 query rows of one head; K/V tiles for
// each 32-key step are staged once in LDS (bf16, fragment-native layout)
// and shared by all 4 waves:
//   S = (Q Kt)/8 + spdE[spd] + edgeE[edge]   (2 WMMAs per 16-key subtile)
//   online softmax over keys (shfl_xor width-16 row reductions)
//   O += P @ V  (P: C-layout -> LDS (A-layout) -> 2x ds_load_b128; 4 WMMAs)
// spd/edge matrices (16.7 MB each) are L2-resident (192 MB L2) -> direct
// global gathers, no LDS staging needed.
// ---------------------------------------------------------------------------
__global__ __launch_bounds__(128)
void attn_kernel(const float* __restrict__ q, const float* __restrict__ k,
                 const float* __restrict__ vv,
                 const int* __restrict__ spd, const int* __restrict__ edge,
                 const float* __restrict__ spdE, const float* __restrict__ edgeE,
                 float* __restrict__ out)
{
    const int h  = blockIdx.y;
    const int hb = h * HDIM;
    const int tid  = threadIdx.x;
    const int wave = tid >> 5;
    const int lane = tid & 31;
    const int i0 = blockIdx.x * 64 + wave * 16;
    const int col  = lane & 15;
    const int half = lane >> 4;
    const int rb   = half * 8;

    __shared__ alignas(16) __bf16 Kt[32][64];       // [key][d]      4 KB
    __shared__ alignas(16) __bf16 Vt[64][32];       // [d][key]      4 KB
    __shared__ alignas(16) __bf16 Ps[4][16][32];    // per-wave P    4 KB

    // Q fragments (rows i0..i0+15): d 0..31 (qa0) and 32..63 (qa1)
    v16bf qa0, qa1;
    {
        const float* qr = &q[(size_t)(i0 + col) * DMODEL + hb];
        int kb = half * 16;
        #pragma unroll
        for (int s = 0; s < 2; ++s) {           // element octet s -> d base kb + s*8
            float4 lo = *reinterpret_cast<const float4*>(&qr[kb + s * 8]);
            float4 hi = *reinterpret_cast<const float4*>(&qr[kb + s * 8 + 4]);
            qa0[s * 8 + 0] = (__bf16)lo.x; qa0[s * 8 + 1] = (__bf16)lo.y;
            qa0[s * 8 + 2] = (__bf16)lo.z; qa0[s * 8 + 3] = (__bf16)lo.w;
            qa0[s * 8 + 4] = (__bf16)hi.x; qa0[s * 8 + 5] = (__bf16)hi.y;
            qa0[s * 8 + 6] = (__bf16)hi.z; qa0[s * 8 + 7] = (__bf16)hi.w;
            float4 lo1 = *reinterpret_cast<const float4*>(&qr[32 + kb + s * 8]);
            float4 hi1 = *reinterpret_cast<const float4*>(&qr[32 + kb + s * 8 + 4]);
            qa1[s * 8 + 0] = (__bf16)lo1.x; qa1[s * 8 + 1] = (__bf16)lo1.y;
            qa1[s * 8 + 2] = (__bf16)lo1.z; qa1[s * 8 + 3] = (__bf16)lo1.w;
            qa1[s * 8 + 4] = (__bf16)hi1.x; qa1[s * 8 + 5] = (__bf16)hi1.y;
            qa1[s * 8 + 6] = (__bf16)hi1.z; qa1[s * 8 + 7] = (__bf16)hi1.w;
        }
    }

    float mrow[8], lrow[8];
    #pragma unroll
    for (int r = 0; r < 8; ++r) { mrow[r] = -1.0e30f; lrow[r] = 0.0f; }
    v8f o0 = {}, o1 = {}, o2 = {}, o3 = {};

    const int pc = (col & 7) + ((col >> 3) << 4);   // apermute(col) for j<16

    for (int j0 = 0; j0 < NTOK; j0 += 32) {
        __syncthreads();    // previous tile fully consumed
        // cooperative staging of K/V tile: 32 keys x 64 d, float4 loads
        #pragma unroll
        for (int t = 0; t < 4; ++t) {
            int g = tid + t * 128;                  // 512 groups of 4
            int key = g >> 4, dg = (g & 15) << 2;
            const float4 kv = *reinterpret_cast<const float4*>(
                &k[(size_t)(j0 + key) * DMODEL + hb + dg]);
            v4bf wk;
            wk[0] = (__bf16)kv.x; wk[1] = (__bf16)kv.y;
            wk[2] = (__bf16)kv.z; wk[3] = (__bf16)kv.w;
            *reinterpret_cast<v4bf*>(&Kt[key][dg]) = wk;
            const float4 vvv = *reinterpret_cast<const float4*>(
                &vv[(size_t)(j0 + key) * DMODEL + hb + dg]);
            Vt[dg + 0][key] = (__bf16)vvv.x;
            Vt[dg + 1][key] = (__bf16)vvv.y;
            Vt[dg + 2][key] = (__bf16)vvv.z;
            Vt[dg + 3][key] = (__bf16)vvv.w;
        }
        __syncthreads();

        float s[2][8];
        #pragma unroll
        for (int f = 0; f < 2; ++f) {
            const int jb = j0 + f * 16;
            // B = K^T fragments from LDS: lane col -> key, 16 contiguous d/half
            const v16bf b0 = *reinterpret_cast<const v16bf*>(&Kt[f * 16 + col][half * 16]);
            const v16bf b1 = *reinterpret_cast<const v16bf*>(&Kt[f * 16 + col][32 + half * 16]);
            v8f sc = {};
            sc = __builtin_amdgcn_wmma_f32_16x16x32_bf16(false, qa0, false, b0, (short)0, sc, false, false);
            sc = __builtin_amdgcn_wmma_f32_16x16x32_bf16(false, qa1, false, b1, (short)0, sc, false, false);
            #pragma unroll
            for (int r = 0; r < 8; ++r) {
                int gi = i0 + rb + r;
                int gj = jb + col;
                int sp = clampi(spd[(size_t)gi * NTOK + gj], 0, 10);
                int ed = clampi(edge[(size_t)gi * NTOK + gj], 0, 5);
                s[f][r] = sc[r] * 0.125f + spdE[sp * HEADS + h] + edgeE[ed * HEADS + h];
            }
        }
        // online softmax over the 32 new keys
        #pragma unroll
        for (int r = 0; r < 8; ++r) {
            float lm = fmaxf(s[0][r], s[1][r]);
            #pragma unroll
            for (int m = 1; m < 16; m <<= 1) lm = fmaxf(lm, __shfl_xor(lm, m, 32));
            float mnew  = fmaxf(mrow[r], lm);
            float scale = __expf(mrow[r] - mnew);
            float p0 = __expf(s[0][r] - mnew);
            float p1 = __expf(s[1][r] - mnew);
            float ls = p0 + p1;
            #pragma unroll
            for (int m = 1; m < 16; m <<= 1) ls += __shfl_xor(ls, m, 32);
            lrow[r] = lrow[r] * scale + ls;
            mrow[r] = mnew;
            o0[r] *= scale; o1[r] *= scale; o2[r] *= scale; o3[r] *= scale;
            Ps[wave][rb + r][col]      = (__bf16)p0;   // j = col
            Ps[wave][rb + r][16 + col] = (__bf16)p1;   // j = 16 + col
        }
        // P as A fragment (own wave slice; wave-internal RAW handled by dscnt)
        const v16bf pa = reinterpret_cast<const v16bf*>(&Ps[wave][col][0])[half];
        // V fragments from LDS (32 keys x 16 head-cols) x 4 -> accumulate O
        const v16bf bv0 = *reinterpret_cast<const v16bf*>(&Vt[col][half * 16]);
        const v16bf bv1 = *reinterpret_cast<const v16bf*>(&Vt[16 + col][half * 16]);
        const v16bf bv2 = *reinterpret_cast<const v16bf*>(&Vt[32 + col][half * 16]);
        const v16bf bv3 = *reinterpret_cast<const v16bf*>(&Vt[48 + col][half * 16]);
        o0 = __builtin_amdgcn_wmma_f32_16x16x32_bf16(false, pa, false, bv0, (short)0, o0, false, false);
        o1 = __builtin_amdgcn_wmma_f32_16x16x32_bf16(false, pa, false, bv1, (short)0, o1, false, false);
        o2 = __builtin_amdgcn_wmma_f32_16x16x32_bf16(false, pa, false, bv2, (short)0, o2, false, false);
        o3 = __builtin_amdgcn_wmma_f32_16x16x32_bf16(false, pa, false, bv3, (short)0, o3, false, false);
    }

    #pragma unroll
    for (int r = 0; r < 8; ++r) {
        int gi = i0 + rb + r;
        float inv = 1.0f / lrow[r];
        out[(size_t)gi * DMODEL + hb + col]      = o0[r] * inv;
        out[(size_t)gi * DMODEL + hb + 16 + col] = o1[r] * inv;
        out[(size_t)gi * DMODEL + hb + 32 + col] = o2[r] * inv;
        out[(size_t)gi * DMODEL + hb + 48 + col] = o3[r] * inv;
    }
}

// ---------------------------------------------------------------------------
// LayerNorm over D=512, one wave per row: lane owns 16 contiguous columns
// (4x float4 loads), full-wave shfl_xor reduction.
// ---------------------------------------------------------------------------
__global__ __launch_bounds__(32)
void layernorm_kernel(const float* __restrict__ in, const float* __restrict__ g,
                      const float* __restrict__ b, float* __restrict__ out)
{
    const int row = blockIdx.x;
    const int lane = threadIdx.x;
    const int c0 = lane * 16;
    float4 xv[4];
    float s = 0.0f, s2 = 0.0f;
    #pragma unroll
    for (int i = 0; i < 4; ++i) {
        xv[i] = *reinterpret_cast<const float4*>(&in[(size_t)row * DMODEL + c0 + i * 4]);
        s  += xv[i].x + xv[i].y + xv[i].z + xv[i].w;
        s2 += xv[i].x * xv[i].x + xv[i].y * xv[i].y + xv[i].z * xv[i].z + xv[i].w * xv[i].w;
    }
    #pragma unroll
    for (int m = 1; m < 32; m <<= 1) { s += __shfl_xor(s, m, 32); s2 += __shfl_xor(s2, m, 32); }
    float mean = s * (1.0f / DMODEL);
    float var  = s2 * (1.0f / DMODEL) - mean * mean;
    float inv  = rsqrtf(var + 1e-5f);
    #pragma unroll
    for (int i = 0; i < 4; ++i) {
        float4 gv = *reinterpret_cast<const float4*>(&g[c0 + i * 4]);
        float4 bv = *reinterpret_cast<const float4*>(&b[c0 + i * 4]);
        float4 o;
        o.x = (xv[i].x - mean) * inv * gv.x + bv.x;
        o.y = (xv[i].y - mean) * inv * gv.y + bv.y;
        o.z = (xv[i].z - mean) * inv * gv.z + bv.z;
        o.w = (xv[i].w - mean) * inv * gv.w + bv.w;
        *reinterpret_cast<float4*>(&out[(size_t)row * DMODEL + c0 + i * 4]) = o;
    }
}

// ---------------------------------------------------------------------------
// logits = x @ fc_w[512,7] + fc_b ; log_softmax over 7. One wave per row.
// ---------------------------------------------------------------------------
__global__ __launch_bounds__(32)
void head_kernel(const float* __restrict__ x, const float* __restrict__ w,
                 const float* __restrict__ b, float* __restrict__ out)
{
    const int row = blockIdx.x;
    const int lane = threadIdx.x;
    float acc[OUTC];
    #pragma unroll
    for (int o = 0; o < OUTC; ++o) acc[o] = 0.0f;
    for (int kk = lane; kk < DMODEL; kk += 32) {
        float xv = x[(size_t)row * DMODEL + kk];
        #pragma unroll
        for (int o = 0; o < OUTC; ++o) acc[o] += xv * w[kk * OUTC + o];
    }
    #pragma unroll
    for (int o = 0; o < OUTC; ++o)
        #pragma unroll
        for (int m = 1; m < 32; m <<= 1) acc[o] += __shfl_xor(acc[o], m, 32);
    if (lane == 0) {
        float z[OUTC]; float mx = -1.0e30f;
        #pragma unroll
        for (int o = 0; o < OUTC; ++o) { z[o] = acc[o] + b[o]; mx = fmaxf(mx, z[o]); }
        float se = 0.0f;
        #pragma unroll
        for (int o = 0; o < OUTC; ++o) se += __expf(z[o] - mx);
        float lse = mx + __logf(se);
        #pragma unroll
        for (int o = 0; o < OUTC; ++o) out[(size_t)row * OUTC + o] = z[o] - lse;
    }
}

// ---------------------------------------------------------------------------
// Host launcher. Input order assumption (jax pytree flattening: top-level
// dict in insertion order, 'params' leaves in sorted-key order, layers list
// in order):
//   0:x 1:spd 2:cin 3:cout 4:edge
//   5:embed_b 6:embed_w 7:fc_b 8:fc_w 9:in_cin 10:in_cout
//   per layer (base=11+l*20):
//   +0 cin +1 cout +2 edge +3 f1b +4 f1w +5 f2b +6 f2w +7 kb +8 kw
//   +9 ln1b +10 ln1g +11 ln2b +12 ln2g +13 ob +14 ow +15 qb +16 qw
//   +17 spd +18 vb +19 vw
// ---------------------------------------------------------------------------
extern "C" void kernel_launch(void* const* d_in, const int* in_sizes, int n_in,
                              void* d_out, int out_size, void* d_ws, size_t ws_size,
                              hipStream_t stream) {
    (void)in_sizes; (void)n_in; (void)out_size; (void)ws_size;

    const float* x_in    = (const float*)d_in[0];
    const int*   spd     = (const int*)d_in[1];
    const int*   cin     = (const int*)d_in[2];
    const int*   cout_   = (const int*)d_in[3];
    const int*   edge    = (const int*)d_in[4];
    const float* embed_b = (const float*)d_in[5];
    const float* embed_w = (const float*)d_in[6];
    const float* fc_b    = (const float*)d_in[7];
    const float* fc_w    = (const float*)d_in[8];
    const float* in_cin  = (const float*)d_in[9];
    const float* in_cout = (const float*)d_in[10];

    float* W = (float*)d_ws;
    const size_t S = (size_t)NTOK * DMODEL;
    float* bx   = W;            // current x
    float* bxc  = W + 1 * S;    // x + layer centrality enc
    float* bq   = W + 2 * S;
    float* bk   = W + 3 * S;
    float* bv   = W + 4 * S;
    float* batt = W + 5 * S;    // attention output pre-projection
    float* bt   = W + 6 * S;    // pre-LN sums
    float* bx1  = W + 7 * S;    // LN1 output
    float* bff  = W + 8 * S;    // gelu(ffn1), NTOK x FFDIM

    const dim3 blk(256);
    const dim3 g512(DMODEL / 64, NTOK / 64);
    const dim3 gff (FFDIM  / 64, NTOK / 64);

    // embed: h = x @ embed_w + embed_b + in_cin[cin] + in_cout[cout]
    gemm_bf16_kernel<<<g512, blk, 0, stream>>>(
        x_in, embed_w, embed_b, bx, nullptr, cin, in_cin, cout_, in_cout,
        NTOK, DMODEL, INDIM, 0);

    for (int l = 0; l < 2; ++l) {
        const int base = 11 + l * 20;
        const float* cinE  = (const float*)d_in[base + 0];
        const float* coutE = (const float*)d_in[base + 1];
        const float* edgeE = (const float*)d_in[base + 2];
        const float* f1b   = (const float*)d_in[base + 3];
        const float* f1w   = (const float*)d_in[base + 4];
        const float* f2b   = (const float*)d_in[base + 5];
        const float* f2w   = (const float*)d_in[base + 6];
        const float* kbv   = (const float*)d_in[base + 7];
        const float* kw    = (const float*)d_in[base + 8];
        const float* ln1b  = (const float*)d_in[base + 9];
        const float* ln1g  = (const float*)d_in[base + 10];
        const float* ln2b  = (const float*)d_in[base + 11];
        const float* ln2g  = (const float*)d_in[base + 12];
        const float* ob    = (const float*)d_in[base + 13];
        const float* ow    = (const float*)d_in[base + 14];
        const float* qbv   = (const float*)d_in[base + 15];
        const float* qw    = (const float*)d_in[base + 16];
        const float* spdE  = (const float*)d_in[base + 17];
        const float* vbv   = (const float*)d_in[base + 18];
        const float* vw    = (const float*)d_in[base + 19];

        add_cent_kernel<<<(NTOK * DMODEL) / 1024, blk, 0, stream>>>(bx, cin, cout_, cinE, coutE, bxc);

        gemm_bf16_kernel<<<g512, blk, 0, stream>>>(bxc, qw, qbv, bq, nullptr, nullptr, nullptr, nullptr, nullptr,
                                                   NTOK, DMODEL, DMODEL, 0);
        gemm_bf16_kernel<<<g512, blk, 0, stream>>>(bxc, kw, kbv, bk, nullptr, nullptr, nullptr, nullptr, nullptr,
                                                   NTOK, DMODEL, DMODEL, 0);
        gemm_bf16_kernel<<<g512, blk, 0, stream>>>(bxc, vw, vbv, bv, nullptr, nullptr, nullptr, nullptr, nullptr,
                                                   NTOK, DMODEL, DMODEL, 0);

        attn_kernel<<<dim3(NTOK / 64, HEADS), 128, 0, stream>>>(bq, bk, bv, spd, edge, spdE, edgeE, batt);

        // out-projection + residual (pre-centrality x)
        gemm_bf16_kernel<<<g512, blk, 0, stream>>>(batt, ow, ob, bt, bx, nullptr, nullptr, nullptr, nullptr,
                                                   NTOK, DMODEL, DMODEL, 0);
        layernorm_kernel<<<NTOK, 32, 0, stream>>>(bt, ln1g, ln1b, bx1);

        // FFN1 + exact GELU
        gemm_bf16_kernel<<<gff, blk, 0, stream>>>(bx1, f1w, f1b, bff, nullptr, nullptr, nullptr, nullptr, nullptr,
                                                  NTOK, FFDIM, DMODEL, 1);
        // FFN2 + residual
        gemm_bf16_kernel<<<g512, blk, 0, stream>>>(bff, f2w, f2b, bt, bx1, nullptr, nullptr, nullptr, nullptr,
                                                   NTOK, DMODEL, FFDIM, 0);
        layernorm_kernel<<<NTOK, 32, 0, stream>>>(bt, ln2g, ln2b, bx);
    }

    head_kernel<<<NTOK, 32, 0, stream>>>(bx, fc_w, fc_b, (float*)d_out);
}
